// DelayedAgg_76690936037739
// MI455X (gfx1250) — matrix operand order repaired and checked by reading, not compiled, not run on hardware
//
#include <hip/hip_runtime.h>

// ---------------------------------------------------------------------------
// Problem constants (match reference)
// ---------------------------------------------------------------------------
#define B_    8
#define N_    24000
#define M_    6000
#define K_    32
#define CIN   64
#define CMID  128
#define COUT  256
#define NPT   96        // support points per MLP workgroup (24000 = 250 * 96)
#define TM    16        // query points per gather workgroup
#define XGS   12        // dwords per LDS operand group: 8 data + 4 pad (48B, 16B-aligned)

typedef __bf16 bf16_t;
typedef __attribute__((ext_vector_type(16))) __bf16    v16bf;
typedef __attribute__((ext_vector_type(2)))  __bf16    v2bf;
typedef __attribute__((ext_vector_type(8)))  float     v8f;
typedef __attribute__((ext_vector_type(4)))  unsigned  v4u;
typedef __attribute__((ext_vector_type(8)))  unsigned  v8u;

// workspace layout (bytes)
#define Y_BYTES   ((size_t)B_ * N_ * COUT * 4)     // 196,608,000 (32B-aligned)
#define W1A_OFF   (Y_BYTES)
#define W1A_BYTES ((size_t)8 * 3 * 32 * 8 * 4)     // 24,576
#define W2A_OFF   (W1A_OFF + W1A_BYTES)            // W2A: 16*4*32*8*4 = 65,536

__device__ __forceinline__ unsigned pack2(float a, float b) {
  v2bf p; p[0] = (bf16_t)a; p[1] = (bf16_t)b;
  return __builtin_bit_cast(unsigned, p);
}

// load one WMMA 16-bit operand (8 contiguous dwords) as two b128 loads
__device__ __forceinline__ v16bf ld_op8(const unsigned* p) {
  v4u lo = *(const v4u*)p;
  v4u hi = *(const v4u*)(p + 4);
  return __builtin_bit_cast(v16bf, __builtin_shufflevector(lo, hi, 0,1,2,3,4,5,6,7));
}

// ---------------------------------------------------------------------------
// Kernel 0: pre-pack weights to bf16 in per-lane WMMA A-operand order.
// A-layout (16-bit 16x32, ISA 7.12.2): lane holds row M=lane%16; K-half by
// lane/16; VGPR v packs K pairs (v<4: kb+2v, else 16+kb+2(v-4)), kb=8*(lane/16).
// Element i = ((row_tile*S + s)*32 + lane)*8 + v  -> one coalesced 32B/lane load.
// ---------------------------------------------------------------------------
__global__ void prep_weights_kernel(const float* __restrict__ W1,
                                    const float* __restrict__ W2,
                                    unsigned* __restrict__ W1a,
                                    unsigned* __restrict__ W2a) {
  const int t = threadIdx.x;
  for (int i = t; i < 8 * 3 * 32 * 8; i += 256) {          // layer 1 (K 67->96 pad)
    int v = i & 7, lane = (i >> 3) & 31, s = (i >> 8) % 3, rt = i / 768;
    int row = rt * 16 + (lane & 15);
    int kb  = (lane >> 4) * 8;
    int K   = s * 32 + ((v < 4) ? (kb + 2 * v) : (16 + kb + 2 * (v - 4)));
    float a = (K     < 67) ? W1[row * 67 + K]     : 0.f;
    float c = (K + 1 < 67) ? W1[row * 67 + K + 1] : 0.f;
    W1a[i] = pack2(a, c);
  }
  for (int i = t; i < 16 * 4 * 32 * 8; i += 256) {         // layer 2 (K = 128)
    int v = i & 7, lane = (i >> 3) & 31, s = (i >> 8) & 3, rt = i >> 10;
    int row = rt * 16 + (lane & 15);
    int kb  = (lane >> 4) * 8;
    int K   = s * 32 + ((v < 4) ? (kb + 2 * v) : (16 + kb + 2 * (v - 4)));
    W2a[i] = pack2(W2[row * CMID + K], W2[row * CMID + K + 1]);
  }
}

// ---------------------------------------------------------------------------
// Kernel 1: fused 2-layer pointwise MLP via v_wmma_f32_16x16x32_bf16.
// LDS holds X (96xNPT) and H (128xNPT) in B-operand group order:
//   group g = Khalf*(S*NPT) + kstep*NPT + column, 8 dwords (+4 pad) per group,
// so each lane's B operand is 8 contiguous dwords -> 2x ds_load_b128, and the
// H writeback is one ds_store_b128 per tile. A operands are single coalesced
// 32B/lane global loads from the pre-packed weights. N%NPT==0 -> no bounds
// checks -> EXEC all-ones throughout (WMMA requirement).
// Output Y is POINT-MAJOR [b][n][256] so the gather stage reads 1KB rows.
// ---------------------------------------------------------------------------
__global__ __launch_bounds__(256) void mlp_wmma_kernel(
    const float* __restrict__ features,   // [B][CIN][N]
    const float* __restrict__ sxyz,       // [B][N][3]
    const float* __restrict__ b1,         // [CMID]
    const float* __restrict__ b2,         // [COUT]
    const unsigned* __restrict__ W1a,     // packed A-operands, layer 1
    const unsigned* __restrict__ W2a,     // packed A-operands, layer 2
    float* __restrict__ Y)                // [B][N][COUT]
{
  __shared__ unsigned Xs3[2 * 3 * NPT * XGS];   // 6912 dw = 27,648 B
  __shared__ unsigned Hs3[2 * 4 * NPT * XGS];   // 9216 dw = 36,864 B

  const int b    = blockIdx.y;
  const int n0   = blockIdx.x * NPT;
  const int t    = threadIdx.x;
  const int lane = t & 31;
  const int w    = t >> 5;           // wave id 0..7
  const int lr   = lane & 15;
  const int hs   = lane >> 4;        // K-half select (B layout)

  // ---- build X in B-operand group order; pt fastest -> coalesced loads ----
  const float* featB = features + (size_t)b * CIN * N_;
  for (int i = t; i < 2 * 3 * 8 * NPT; i += 256) {
    int pt = i % NPT;
    int r  = i / NPT;                // r = (hh*3 + s)*8 + v
    int v  = r & 7;
    int s  = (r >> 3) % 3;
    int hh = r / 24;
    int K  = s * 32 + 8 * hh + ((v < 4) ? 2 * v : 16 + 2 * (v - 4));   // even K of the packed pair
    int n  = n0 + pt;
    float a, c;
    if (K < 64) {
      a = featB[(size_t)K * N_ + n];
      c = featB[(size_t)(K + 1) * N_ + n];
    } else if (K == 64) {
      const float* p = sxyz + ((size_t)b * N_ + n) * 3;
      a = p[0]; c = p[1];
    } else if (K == 66) {
      a = sxyz[((size_t)b * N_ + n) * 3 + 2]; c = 0.f;
    } else {
      a = 0.f; c = 0.f;
    }
    Xs3[(hh * (3 * NPT) + s * NPT + pt) * XGS + v] = pack2(a, c);
  }
  __syncthreads();

  // ---- layer 1: H = relu(W1*X + b1); wave w -> output rows 16w..16w+15 ----
  {
    v16bf A[3];
    #pragma unroll
    for (int s = 0; s < 3; ++s)
      A[s] = ld_op8(W1a + ((w * 3 + s) * 32 + lane) * 8);

    const int ch0 = 16 * w + 8 * hs;           // acc[i] -> channel ch0+i
    const int s2  = ch0 >> 5;                  // H-group coords for writeback
    const int kk  = ch0 & 31;
    const int hs2 = (kk >> 3) & 1;
    const int vb  = (kk >> 4) * 4;             // 0 or 4 -> 16B-aligned b128
    float bias[8];
    #pragma unroll
    for (int i = 0; i < 8; ++i) bias[i] = b1[ch0 + i];

    #pragma unroll
    for (int c = 0; c < 6; ++c) {
      const int pt = 16 * c + lr;
      v8f acc = {};
      #pragma unroll
      for (int s = 0; s < 3; ++s) {
        v16bf Bv = ld_op8(Xs3 + (hs * (3 * NPT) + s * NPT + pt) * XGS);
        acc = __builtin_amdgcn_wmma_f32_16x16x32_bf16(
                  false, A[s], false, Bv, (short)0, acc, false, false);
      }
      v4u hq;
      #pragma unroll
      for (int j = 0; j < 4; ++j)
        hq[j] = pack2(fmaxf(acc[2 * j]     + bias[2 * j],     0.f),
                      fmaxf(acc[2 * j + 1] + bias[2 * j + 1], 0.f));
      *(v4u*)(Hs3 + (hs2 * (4 * NPT) + s2 * NPT + pt) * XGS + vb) = hq;
    }
  }
  __syncthreads();

  // ---- layer 2: Y = relu(W2*H + b2); wave w -> row-tiles 2w, 2w+1 ----
  #pragma unroll
  for (int rt = 2 * w; rt <= 2 * w + 1; ++rt) {
    v16bf A[4];
    #pragma unroll
    for (int s = 0; s < 4; ++s)
      A[s] = ld_op8(W2a + ((rt * 4 + s) * 32 + lane) * 8);

    const int ch0 = 16 * rt + 8 * hs;
    float bias[8];
    #pragma unroll
    for (int i = 0; i < 8; ++i) bias[i] = b2[ch0 + i];

    #pragma unroll
    for (int c = 0; c < 6; ++c) {
      const int pt = 16 * c + lr;
      v8f acc = {};
      #pragma unroll
      for (int s = 0; s < 4; ++s) {
        v16bf Bv = ld_op8(Hs3 + (hs * (4 * NPT) + s * NPT + pt) * XGS);
        acc = __builtin_amdgcn_wmma_f32_16x16x32_bf16(
                  false, A[s], false, Bv, (short)0, acc, false, false);
      }
      float4 f0 = {fmaxf(acc[0] + bias[0], 0.f), fmaxf(acc[1] + bias[1], 0.f),
                   fmaxf(acc[2] + bias[2], 0.f), fmaxf(acc[3] + bias[3], 0.f)};
      float4 f1 = {fmaxf(acc[4] + bias[4], 0.f), fmaxf(acc[5] + bias[5], 0.f),
                   fmaxf(acc[6] + bias[6], 0.f), fmaxf(acc[7] + bias[7], 0.f)};
      float* dst = Y + ((size_t)b * N_ + n0 + pt) * COUT + ch0;   // 32B aligned
      *(float4*)(dst)     = f0;
      *(float4*)(dst + 4) = f1;
    }
  }
}

// ---------------------------------------------------------------------------
// Kernel 2: neighbor gather + max reduce.
// Thread t owns channel t (COUT == 256 == blockDim). Each neighbor read is a
// fully coalesced 1KB row (point-major Y) that stays L2-resident
// (24.6 MB/batch << 192 MB L2). Output written as 16 contiguous m per thread.
// ---------------------------------------------------------------------------
__global__ __launch_bounds__(256) void gather_max_kernel(
    const int* __restrict__ nidx,     // [B][M][K]
    const float* __restrict__ Y,      // [B][N][COUT]
    float* __restrict__ out)          // [B][COUT][M]
{
  const int b  = blockIdx.y;
  const int m0 = blockIdx.x * TM;
  const int t  = threadIdx.x;         // channel

  __shared__ int idxs[TM * K_];
  for (int i = t; i < TM * K_; i += 256)
    idxs[i] = nidx[((size_t)b * M_ + m0) * K_ + i];
  __syncthreads();

  const float* Yb = Y + (size_t)b * N_ * COUT;
  float acc[TM];
  #pragma unroll
  for (int j = 0; j < TM; ++j) {
    if (j + 1 < TM) {  // warm next query's neighbor rows (global_prefetch_b8)
      int kpf = (t >> 3) & (K_ - 1);
      __builtin_prefetch(Yb + (size_t)idxs[(j + 1) * K_ + kpf] * COUT + t, 0, 1);
    }
    float a = -INFINITY;
    #pragma unroll
    for (int k = 0; k < K_; ++k)
      a = fmaxf(a, Yb[(size_t)idxs[j * K_ + k] * COUT + t]);
    acc[j] = a;
  }

  float* dst = out + ((size_t)b * COUT + t) * M_ + m0;  // 16B aligned (m0%16==0)
  float4* d4 = (float4*)dst;
  #pragma unroll
  for (int j = 0; j < TM / 4; ++j) {
    float4 f = {acc[4 * j], acc[4 * j + 1], acc[4 * j + 2], acc[4 * j + 3]};
    d4[j] = f;
  }
}

// ---------------------------------------------------------------------------
extern "C" void kernel_launch(void* const* d_in, const int* in_sizes, int n_in,
                              void* d_out, int out_size, void* d_ws, size_t ws_size,
                              hipStream_t stream) {
  // setup_inputs order: query_xyz, support_xyz, features, neighbor_idx, W1, b1, W2, b2
  const float* sxyz     = (const float*)d_in[1];
  const float* features = (const float*)d_in[2];
  const int*   nidx     = (const int*)  d_in[3];
  const float* W1       = (const float*)d_in[4];
  const float* b1       = (const float*)d_in[5];
  const float* W2       = (const float*)d_in[6];
  const float* b2       = (const float*)d_in[7];
  float* out = (float*)d_out;

  char* ws = (char*)d_ws;
  float*    Y   = (float*)ws;
  unsigned* W1a = (unsigned*)(ws + W1A_OFF);
  unsigned* W2a = (unsigned*)(ws + W2A_OFF);

  prep_weights_kernel<<<dim3(1), dim3(256), 0, stream>>>(W1, W2, W1a, W2a);

  mlp_wmma_kernel<<<dim3(N_ / NPT, B_), dim3(256), 0, stream>>>(
      features, sxyz, b1, b2, W1a, W2a, Y);

  gather_max_kernel<<<dim3(M_ / TM, B_), dim3(256), 0, stream>>>(nidx, Y, out);
}
